// GraphNetworkLayerWithCoords_29669634081214
// MI455X (gfx1250) — matrix operand order repaired and checked by reading, not compiled, not run on hardware
//
#include <hip/hip_runtime.h>
#include <hip/hip_bf16.h>

typedef __attribute__((ext_vector_type(16))) _Float16 v16h;
typedef __attribute__((ext_vector_type(8)))  _Float16 v8h;
typedef __attribute__((ext_vector_type(4)))  _Float16 v4h;
typedef __attribute__((ext_vector_type(8)))  float    v8f;

#define NNODES 50000
#define NEDGES 800000
#define HDIM   128
#define EFDIM  32

// LDS strides (elements), chosen multiple-of-8 halves for 16B-aligned ds_load_b128
#define MSG_STRIDE 328   // halves, K padded 291 -> 320 (+8 pad vs bank conflicts)
#define H1_STRIDE  136   // halves, K = 128
#define UPD_STRIDE 264   // halves, K = 256
#define OUT_STRIDE 129   // floats (conflict-free row reads in layernorm)

static __device__ __forceinline__ v8f wmma_f16(v16h a, v16h b, v8f c) {
  // D = A(16x32 f16) * B(32x16 f16) + C(16x16 f32)
  return __builtin_amdgcn_wmma_f32_16x16x32_f16(false, a, false, b, (short)0, c,
                                                false, false);
}

// A-fragment per ISA 7.12.2 (16-bit A 16x32): lane m = lane&15, hi = lane>>4
//   elems 0..7  -> K = k0 + 8*hi + e
//   elems 8..15 -> K = k0 + 16 + 8*hi + (e-8)
static __device__ __forceinline__ v16h load_a_frag(const _Float16* base,
                                                   int strideHalves, int row0,
                                                   int k0, int lane) {
  const int hi = lane >> 4;
  const int m  = lane & 15;
  const _Float16* p = base + (row0 + m) * strideHalves + k0 + hi * 8;
  v8h lo = *(const v8h*)p;
  v8h hh = *(const v8h*)(p + 16);
  v16h a;
#pragma unroll
  for (int i = 0; i < 8; ++i) { a[i] = lo[i]; a[i + 8] = hh[i]; }
  return a;
}

// Batched-B GEMM step: load all 8 B fragments of K-chunk kc as one clause,
// then 8 back-to-back WMMAs (avoids a full loadcnt-0 wait per WMMA).
// wlane = weights + (lane<<4); fragment (kc,nt) sits at immediate offset
// (kc*8+nt)*512 halves -> global_load_b128 with immediate offsets.
static __device__ __forceinline__ void gemm_chunk(const _Float16* __restrict__ wlane,
                                                  int kc, v16h a, v8f acc[8]) {
  v16h bfr[8];
#pragma unroll
  for (int nt = 0; nt < 8; ++nt)
    bfr[nt] = *(const v16h*)(wlane + (kc * 8 + nt) * 512);
#pragma unroll
  for (int nt = 0; nt < 8; ++nt)
    acc[nt] = wmma_f16(a, bfr[nt], acc[nt]);
}

// ---------------------------------------------------------------------------
// Pre-swizzle fp32 row-major W[K][128] into per-lane-contiguous f16 B-fragments:
// b[e] of lane L (n = L&15, hi = L>>4) = W[kc*32 + 16*(e>>3) + 8*hi + (e&7)][nt*16 + n]
// storage: out[(((kc*8)+nt)*32 + lane)*16 + e]
// ---------------------------------------------------------------------------
__global__ void swizzle_w_kernel(const float* __restrict__ W,
                                 _Float16* __restrict__ out, int Kreal, int nkc) {
  const int total = nkc * 4096;
  for (int idx = blockIdx.x * blockDim.x + threadIdx.x; idx < total;
       idx += gridDim.x * blockDim.x) {
    const int e    = idx & 15;
    const int lane = (idx >> 4) & 31;
    const int nt   = (idx >> 9) & 7;
    const int kc   = idx >> 12;
    const int k = kc * 32 + ((e >> 3) << 4) + ((lane >> 4) << 3) + (e & 7);
    const int n = nt * 16 + (lane & 15);
    const float v = (k < Kreal) ? W[k * HDIM + n] : 0.0f;
    out[idx] = (_Float16)v;
  }
}

// ---------------------------------------------------------------------------
// Edge kernel: 128 edges per block. Build [x_src | x_dst | eattr | rel | 0pad]
// f16 tile in LDS, GEMM1 (K=320) -> ReLU -> GEMM2 (K=128) -> f32 atomics to aggr.
// Each wave owns rows [16w,16w+16): build/read/write only its strip => no barriers.
// ---------------------------------------------------------------------------
__global__ __launch_bounds__(256) void edge_kernel(
    const float* __restrict__ x, const int* __restrict__ eidx,
    const float* __restrict__ eattr, const float* __restrict__ coords,
    const _Float16* __restrict__ w1s, const float* __restrict__ b1,
    const _Float16* __restrict__ w2s, const float* __restrict__ b2,
    float* __restrict__ aggr) {
  extern __shared__ __align__(16) char smem[];
  _Float16* mt = (_Float16*)smem;                                  // 128 x MSG_STRIDE
  _Float16* h1 = (_Float16*)(smem + 128 * MSG_STRIDE * 2);         // 128 x H1_STRIDE
  int* dsts    = (int*)(smem + 128 * MSG_STRIDE * 2 + 128 * H1_STRIDE * 2);

  const int t     = threadIdx.x;
  const int ebase = blockIdx.x * 128;

  // ---- build tile: thread t -> row t>>1 (belongs to wave t>>5's strip), half p
  {
    const int r = t >> 1, p = t & 1;
    const int e = ebase + r;
    const int s = eidx[e];
    const int d = eidx[NEDGES + e];
    _Float16* row = mt + r * MSG_STRIDE;
    const float4* xs = (const float4*)(x + (size_t)s * HDIM + p * 64);
    const float4* xd = (const float4*)(x + (size_t)d * HDIM + p * 64);
#pragma unroll
    for (int j = 0; j < 16; ++j) {
      float4 v = xs[j];
      v4h h; h[0] = (_Float16)v.x; h[1] = (_Float16)v.y;
             h[2] = (_Float16)v.z; h[3] = (_Float16)v.w;
      *(v4h*)(row + p * 64 + j * 4) = h;
    }
#pragma unroll
    for (int j = 0; j < 16; ++j) {
      float4 v = xd[j];
      v4h h; h[0] = (_Float16)v.x; h[1] = (_Float16)v.y;
             h[2] = (_Float16)v.z; h[3] = (_Float16)v.w;
      *(v4h*)(row + 128 + p * 64 + j * 4) = h;
    }
    const float4* ea = (const float4*)(eattr + (size_t)e * EFDIM + p * 16);
#pragma unroll
    for (int j = 0; j < 4; ++j) {
      float4 v = ea[j];
      v4h h; h[0] = (_Float16)v.x; h[1] = (_Float16)v.y;
             h[2] = (_Float16)v.z; h[3] = (_Float16)v.w;
      *(v4h*)(row + 256 + p * 16 + j * 4) = h;
    }
    if (p == 0) {
      row[288] = (_Float16)(coords[d * 3 + 0] - coords[s * 3 + 0]);
      row[289] = (_Float16)(coords[d * 3 + 1] - coords[s * 3 + 1]);
      row[290] = (_Float16)(coords[d * 3 + 2] - coords[s * 3 + 2]);
#pragma unroll
      for (int j = 291; j < 320; ++j) row[j] = (_Float16)0.0f;
      dsts[r] = d;
    }
  }
  // no __syncthreads(): each wave consumes only rows it built (same-wave DS order)

  const int wv = t >> 5, lane = t & 31, hi = lane >> 4, nl = lane & 15;
  const int row0 = wv * 16;
  const _Float16* w1l = w1s + (lane << 4);
  const _Float16* w2l = w2s + (lane << 4);
  const v8f vzero = {};

  v8f acc[8];
#pragma unroll
  for (int i = 0; i < 8; ++i) acc[i] = vzero;

  // GEMM1: (16 x 320) x (320 x 128)
#pragma unroll
  for (int kc = 0; kc < 10; ++kc) {
    v16h a = load_a_frag(mt, MSG_STRIDE, row0, kc * 32, lane);
    gemm_chunk(w1l, kc, a, acc);
  }
  // bias + ReLU -> h1 tile (f16)
#pragma unroll
  for (int nt = 0; nt < 8; ++nt) {
    const int col = nt * 16 + nl;
    const float bb = b1[col];
#pragma unroll
    for (int r = 0; r < 8; ++r) {
      float h = acc[nt][r] + bb;
      h = h > 0.0f ? h : 0.0f;
      h1[(row0 + r + 8 * hi) * H1_STRIDE + col] = (_Float16)h;
    }
    acc[nt] = vzero;
  }
  // GEMM2: (16 x 128) x (128 x 128)
#pragma unroll
  for (int kc = 0; kc < 4; ++kc) {
    v16h a = load_a_frag(h1, H1_STRIDE, row0, kc * 32, lane);
    gemm_chunk(w2l, kc, a, acc);
  }
  // segment-sum via f32 atomics into L2-resident aggr
#pragma unroll
  for (int nt = 0; nt < 8; ++nt) {
    const int col = nt * 16 + nl;
    const float bb = b2[col];
#pragma unroll
    for (int r = 0; r < 8; ++r) {
      const int d = dsts[row0 + r + 8 * hi];
      atomicAdd(aggr + (size_t)d * HDIM + col, acc[nt][r] + bb);
    }
  }
}

// ---------------------------------------------------------------------------
// Node kernel: 128 nodes per block. [x | aggr] -> GEMM1 -> ReLU -> GEMM2
// -> +bias +residual -> LayerNorm. Out-tile overlays the upd-tile in LDS.
// ---------------------------------------------------------------------------
__global__ __launch_bounds__(256) void node_kernel(
    const float* __restrict__ x, const float* __restrict__ aggr,
    const _Float16* __restrict__ w1s, const float* __restrict__ b1,
    const _Float16* __restrict__ w2s, const float* __restrict__ b2,
    const float* __restrict__ gamma, const float* __restrict__ beta,
    float* __restrict__ out) {
  extern __shared__ __align__(16) char smem[];
  _Float16* ut = (_Float16*)smem;                          // 128 x UPD_STRIDE halves
  float*    ot = (float*)smem;                             // overlay: 128 x OUT_STRIDE f32
  _Float16* h1 = (_Float16*)(smem + 128 * UPD_STRIDE * 2); // 128 x H1_STRIDE

  const int t = threadIdx.x;
  const int nbase = blockIdx.x * 128;

  {
    const int r = t >> 1, p = t & 1;
    const int n = nbase + r;
    _Float16* row = ut + r * UPD_STRIDE;
    if (n < NNODES) {
      const float4* xs = (const float4*)(x + (size_t)n * HDIM + p * 64);
      const float4* as = (const float4*)(aggr + (size_t)n * HDIM + p * 64);
#pragma unroll
      for (int j = 0; j < 16; ++j) {
        float4 v = xs[j];
        v4h h; h[0] = (_Float16)v.x; h[1] = (_Float16)v.y;
               h[2] = (_Float16)v.z; h[3] = (_Float16)v.w;
        *(v4h*)(row + p * 64 + j * 4) = h;
      }
#pragma unroll
      for (int j = 0; j < 16; ++j) {
        float4 v = as[j];
        v4h h; h[0] = (_Float16)v.x; h[1] = (_Float16)v.y;
               h[2] = (_Float16)v.z; h[3] = (_Float16)v.w;
        *(v4h*)(row + 128 + p * 64 + j * 4) = h;
      }
    } else {
      const v4h z = {};
#pragma unroll
      for (int j = 0; j < 16; ++j) {
        *(v4h*)(row + p * 64 + j * 4) = z;
        *(v4h*)(row + 128 + p * 64 + j * 4) = z;
      }
    }
  }

  const int wv = t >> 5, lane = t & 31, hi = lane >> 4, nl = lane & 15;
  const int row0 = wv * 16;
  const _Float16* w1l = w1s + (lane << 4);
  const _Float16* w2l = w2s + (lane << 4);
  const v8f vzero = {};

  v8f acc[8];
#pragma unroll
  for (int i = 0; i < 8; ++i) acc[i] = vzero;

  // GEMM1: (16 x 256) x (256 x 128)
#pragma unroll
  for (int kc = 0; kc < 8; ++kc) {
    v16h a = load_a_frag(ut, UPD_STRIDE, row0, kc * 32, lane);
    gemm_chunk(w1l, kc, a, acc);
  }
#pragma unroll
  for (int nt = 0; nt < 8; ++nt) {
    const int col = nt * 16 + nl;
    const float bb = b1[col];
#pragma unroll
    for (int r = 0; r < 8; ++r) {
      float h = acc[nt][r] + bb;
      h = h > 0.0f ? h : 0.0f;
      h1[(row0 + r + 8 * hi) * H1_STRIDE + col] = (_Float16)h;
    }
    acc[nt] = vzero;
  }
  // GEMM2: (16 x 128) x (128 x 128)
#pragma unroll
  for (int kc = 0; kc < 4; ++kc) {
    v16h a = load_a_frag(h1, H1_STRIDE, row0, kc * 32, lane);
    gemm_chunk(w2l, kc, a, acc);
  }

  __syncthreads();  // all waves done reading ut before overlay writes

  // bias + residual -> f32 out tile
#pragma unroll
  for (int nt = 0; nt < 8; ++nt) {
    const int col = nt * 16 + nl;
    const float bb = b2[col];
#pragma unroll
    for (int r = 0; r < 8; ++r) {
      const int rr = row0 + r + 8 * hi;
      const int n = nbase + rr;
      float v = acc[nt][r] + bb;
      if (n < NNODES) v += x[(size_t)n * HDIM + col];
      ot[rr * OUT_STRIDE + col] = v;
    }
  }

  __syncthreads();

  // LayerNorm: one row per thread (threads 0..127)
  if (t < 128) {
    const int n = nbase + t;
    if (n < NNODES) {
      const float* rowp = ot + t * OUT_STRIDE;
      float s = 0.0f, s2 = 0.0f;
#pragma unroll 8
      for (int i = 0; i < HDIM; ++i) {
        float v = rowp[i];
        s += v;
        s2 += v * v;
      }
      const float mu  = s * (1.0f / HDIM);
      const float var = s2 * (1.0f / HDIM) - mu * mu;
      const float rs  = rsqrtf(var + 1e-5f);
#pragma unroll 8
      for (int i = 0; i < HDIM; ++i) {
        out[(size_t)n * HDIM + i] = (rowp[i] - mu) * rs * gamma[i] + beta[i];
      }
    }
  }
}

// ---------------------------------------------------------------------------
extern "C" void kernel_launch(void* const* d_in, const int* in_sizes, int n_in,
                              void* d_out, int out_size, void* d_ws,
                              size_t ws_size, hipStream_t stream) {
  const float* x      = (const float*)d_in[0];
  const int*   eidx   = (const int*)d_in[1];
  const float* eattr  = (const float*)d_in[2];
  const float* coords = (const float*)d_in[3];
  const float* mW1    = (const float*)d_in[4];
  const float* mb1    = (const float*)d_in[5];
  const float* mW2    = (const float*)d_in[6];
  const float* mb2    = (const float*)d_in[7];
  const float* uW1    = (const float*)d_in[8];
  const float* ub1    = (const float*)d_in[9];
  const float* uW2    = (const float*)d_in[10];
  const float* ub2    = (const float*)d_in[11];
  const float* gamma  = (const float*)d_in[12];
  const float* beta   = (const float*)d_in[13];
  float* out = (float*)d_out;

  char* ws = (char*)d_ws;
  float* aggr = (float*)ws;
  size_t off = (size_t)NNODES * HDIM * sizeof(float);   // 25.6 MB, 256B aligned
  _Float16* mW1s = (_Float16*)(ws + off); off += (size_t)10 * 4096 * 2;
  _Float16* mW2s = (_Float16*)(ws + off); off += (size_t)4  * 4096 * 2;
  _Float16* uW1s = (_Float16*)(ws + off); off += (size_t)8  * 4096 * 2;
  _Float16* uW2s = (_Float16*)(ws + off); off += (size_t)4  * 4096 * 2;

  hipMemsetAsync(aggr, 0, (size_t)NNODES * HDIM * sizeof(float), stream);

  swizzle_w_kernel<<<160, 256, 0, stream>>>(mW1, mW1s, 291, 10);
  swizzle_w_kernel<<<64,  256, 0, stream>>>(mW2, mW2s, 128, 4);
  swizzle_w_kernel<<<128, 256, 0, stream>>>(uW1, uW1s, 256, 8);
  swizzle_w_kernel<<<64,  256, 0, stream>>>(uW2, uW2s, 128, 4);

  const size_t lds_e = 128 * MSG_STRIDE * 2 + 128 * H1_STRIDE * 2 + 128 * 4;
  edge_kernel<<<NEDGES / 128, 256, lds_e, stream>>>(x, eidx, eattr, coords, mW1s,
                                                    mb1, mW2s, mb2, aggr);

  const size_t lds_n = 128 * UPD_STRIDE * 2 + 128 * H1_STRIDE * 2;
  node_kernel<<<(NNODES + 127) / 128, 256, lds_n, stream>>>(
      x, aggr, uW1s, ub1, uW2s, ub2, gamma, beta, out);
}
